// DGC_GraphProp_4475355922590
// MI455X (gfx1250) — matrix operand order repaired and checked by reading, not compiled, not run on hardware
//
#include <hip/hip_runtime.h>
#include <hip/hip_bf16.h>
#include <math.h>

typedef __attribute__((ext_vector_type(16))) __bf16 v16bf;
typedef __attribute__((ext_vector_type(8)))  float  v8f;

#define HID      128
#define N_GRAPHS 64
#define OUT_DIM  32
#define G_DIM    384   // 3*HID
#define H1_DIM   192   // 3*HID/2
#define EPS      0.1f

// ---------------------------------------------------------------------------
// 1) Embedding GEMM: h[n,128] = x[n,128] @ W[128,128] + b   (bf16 WMMA, f32 acc)
// 256 threads = 8 wave32; block owns a 16-row M tile; wave w owns N-tile w.
// A tile staged in LDS as bf16 (one conversion, shared by all 8 waves).
// K loop fully unrolled -> 4 static v_wmma per kernel.
// ---------------------------------------------------------------------------
__global__ __launch_bounds__(256) void k_emb_gemm(
    const float* __restrict__ x, const float* __restrict__ W,
    const float* __restrict__ bias, float* __restrict__ h, int n)
{
    __shared__ __bf16 As[16 * HID];  // 4 KB bf16 A tile
    const int tid  = threadIdx.x;
    const int row0 = blockIdx.x * 16;

    for (int i = tid; i < 16 * HID; i += 256) {
        const int r = row0 + (i >> 7);
        As[i] = (__bf16)((r < n) ? x[(long)r * HID + (i & 127)] : 0.0f);
    }
    __syncthreads();

    const int wave = tid >> 5;
    const int lane = tid & 31;
    const int half = lane >> 4;    // which 16-lane half
    const int lrow = lane & 15;
    const int col  = wave * 16 + lrow;

    v8f acc = {};
#pragma unroll
    for (int kk = 0; kk < HID; kk += 32) {
        v16bf a, b;
#pragma unroll
        for (int e = 0; e < 16; ++e) {
            // A operand, contiguous k order chosen here (matches B): lanes 0-15 hold K 0-15, lanes 16-31 K 16-31
            const int ka = kk + e + half * 16;
            a[e] = As[lrow * HID + ka];
            // B operand, same contiguous k order as A: lanes 0-15 hold K 0-15, lanes 16-31 K 16-31
            const int kb = kk + e + half * 16;
            b[e] = (__bf16)W[(long)kb * HID + col];
        }
        acc = __builtin_amdgcn_wmma_f32_16x16x32_bf16(false, a, false, b,
                                                      (short)0, acc, false, false);
    }

    const float bv = bias[col];
    if (row0 + 16 <= n) {
        // fast path: full tile, no per-row guards (the common case)
#pragma unroll
        for (int v = 0; v < 8; ++v) {
            const int r = row0 + half * 8 + v;   // C/D layout: M = v + half*8
            h[(long)r * HID + col] = acc[v] + bv;
        }
    } else {
#pragma unroll
        for (int v = 0; v < 8; ++v) {
            const int r = row0 + half * 8 + v;
            if (r < n) h[(long)r * HID + col] = acc[v] + bv;
        }
    }
}

// ---------------------------------------------------------------------------
// 2) gcn_norm (improved=True): deg = 2 (self loop) + sum of incoming edges
// ---------------------------------------------------------------------------
__global__ void k_deg_init(float* __restrict__ deg, int n)
{
    const int i = blockIdx.x * blockDim.x + threadIdx.x;
    if (i < n) deg[i] = 2.0f;
}

__global__ void k_deg_edges(const int* __restrict__ col, float* __restrict__ deg, int E)
{
    const int e = blockIdx.x * blockDim.x + threadIdx.x;
    if (e < E) atomicAdd(&deg[col[e]], 1.0f);
}

__global__ void k_dis(const float* __restrict__ deg, float* __restrict__ dis, int n)
{
    const int i = blockIdx.x * blockDim.x + threadIdx.x;
    if (i < n) {
        const float d = deg[i];
        dis[i] = (d > 0.0f) ? __frsqrt_rn(fmaxf(d, 1e-30f)) : 0.0f;
    }
}

// ---------------------------------------------------------------------------
// 3) Propagation: agg = Â h ; h -= eps*agg   (self-loop term seeds agg)
// ---------------------------------------------------------------------------
__global__ void k_prop_init(const float* __restrict__ h, const float* __restrict__ dis,
                            float* __restrict__ agg, int n)
{
    const int t = blockIdx.x * blockDim.x + threadIdx.x;   // n*32 threads
    if (t >= n * 32) return;
    const int i = t >> 5;
    const int d = (t & 31) * 4;
    float s = dis[i];
    s = 2.0f * s * s;                                      // w=2 self loop
    const float4 hv = *(const float4*)(h + (long)i * HID + d);
    float4 o; o.x = s * hv.x; o.y = s * hv.y; o.z = s * hv.z; o.w = s * hv.w;
    *(float4*)(agg + (long)i * HID + d) = o;
}

__global__ void k_prop_edges(const int* __restrict__ row, const int* __restrict__ col,
                             const float* __restrict__ dis, const float* __restrict__ h,
                             float* __restrict__ agg, int E)
{
    const int t = blockIdx.x * blockDim.x + threadIdx.x;   // E*32 threads, 1 wave/edge
    if (t >= E * 32) return;
    const int e = t >> 5;
    const int d = (t & 31) * 4;
    const int r = row[e];
    const int c = col[e];
    const float nrm = dis[r] * dis[c];                     // w=1 edges
    const float4 hv = *(const float4*)(h + (long)r * HID + d);
    float* ap = agg + (long)c * HID + d;                   // agg fits in L2 -> fast atomics
    atomicAdd(ap + 0, nrm * hv.x);
    atomicAdd(ap + 1, nrm * hv.y);
    atomicAdd(ap + 2, nrm * hv.z);
    atomicAdd(ap + 3, nrm * hv.w);
}

__global__ void k_update(float* __restrict__ h, const float* __restrict__ agg,
                         int n, int do_tanh)
{
    const int t = blockIdx.x * blockDim.x + threadIdx.x;   // n*32 threads
    if (t >= n * 32) return;
    const long idx = (long)t * 4;
    float4 hv = *(float4*)(h + idx);
    const float4 av = *(const float4*)(agg + idx);
    hv.x -= EPS * av.x; hv.y -= EPS * av.y; hv.z -= EPS * av.z; hv.w -= EPS * av.w;
    if (do_tanh) {
        hv.x = tanhf(hv.x); hv.y = tanhf(hv.y); hv.z = tanhf(hv.z); hv.w = tanhf(hv.w);
    }
    *(float4*)(h + idx) = hv;
}

// ---------------------------------------------------------------------------
// 4) Tri-pooling (sum / max / mean per graph)
// ---------------------------------------------------------------------------
__device__ inline void atomicMaxF(float* addr, float val)
{
    int old = __float_as_int(*addr);
    while (__int_as_float(old) < val) {
        const int assumed = old;
        old = atomicCAS((int*)addr, assumed, __float_as_int(val));
        if (old == assumed) break;
    }
}

__global__ void k_pool_init(float* __restrict__ sum_p, float* __restrict__ max_p,
                            float* __restrict__ cnt)
{
    const int t = blockIdx.x * blockDim.x + threadIdx.x;
    if (t < N_GRAPHS * HID) { sum_p[t] = 0.0f; max_p[t] = -__builtin_inff(); }
    if (t < N_GRAPHS) cnt[t] = 0.0f;
}

__global__ void k_pool(const float* __restrict__ h, const int* __restrict__ batch,
                       float* __restrict__ sum_p, float* __restrict__ max_p,
                       float* __restrict__ cnt, int n)
{
    const int t = blockIdx.x * blockDim.x + threadIdx.x;   // n*128 threads
    if (t >= n * HID) return;
    const int i = t >> 7;
    const int d = t & 127;
    const int g = batch[i];
    const float v = h[(long)i * HID + d];
    atomicAdd(&sum_p[g * HID + d], v);
    atomicMaxF(&max_p[g * HID + d], v);
    if (d == 0) atomicAdd(&cnt[g], 1.0f);
}

// ---------------------------------------------------------------------------
// 5) MLP head: g=[sum,max,mean] (384) -> leaky(gW1+b1) (192) -> leaky(·W2+b2) (32)
//    One block per graph; everything in LDS. Tiny (~10.2 MFLOP total), scalar OK.
// ---------------------------------------------------------------------------
__global__ __launch_bounds__(H1_DIM) void k_mlp(
    const float* __restrict__ sum_p, const float* __restrict__ max_p,
    const float* __restrict__ cnt,
    const float* __restrict__ W1, const float* __restrict__ b1,
    const float* __restrict__ W2, const float* __restrict__ b2,
    float* __restrict__ out)
{
    __shared__ float gv[G_DIM];
    __shared__ float h1[H1_DIM];
    const int g   = blockIdx.x;
    const int tid = threadIdx.x;
    const float c = fmaxf(cnt[g], 1.0f);

    for (int i = tid; i < G_DIM; i += H1_DIM) {
        float v;
        if (i < HID) {
            v = sum_p[g * HID + i];
        } else if (i < 2 * HID) {
            const float m = max_p[g * HID + (i - HID)];
            v = (m < -3.0e38f) ? 0.0f : m;     // isfinite() guard for empty graphs
        } else {
            v = sum_p[g * HID + (i - 2 * HID)] / c;
        }
        gv[i] = v;
    }
    __syncthreads();

    {   // layer 1
        float acc = b1[tid];
        for (int k = 0; k < G_DIM; ++k) acc = fmaf(gv[k], W1[k * H1_DIM + tid], acc);
        h1[tid] = (acc >= 0.0f) ? acc : 0.01f * acc;
    }
    __syncthreads();

    if (tid < OUT_DIM) {   // layer 2
        float acc = b2[tid];
        for (int k = 0; k < H1_DIM; ++k) acc = fmaf(h1[k], W2[k * OUT_DIM + tid], acc);
        out[g * OUT_DIM + tid] = (acc >= 0.0f) ? acc : 0.01f * acc;
    }
}

// ---------------------------------------------------------------------------
// Host-side orchestration
// ---------------------------------------------------------------------------
extern "C" void kernel_launch(void* const* d_in, const int* in_sizes, int n_in,
                              void* d_out, int out_size, void* d_ws, size_t ws_size,
                              hipStream_t stream)
{
    const float* x     = (const float*)d_in[0];
    const int*   eidx  = (const int*)  d_in[1];
    const int*   batch = (const int*)  d_in[2];
    const float* emb_W = (const float*)d_in[3];
    const float* emb_b = (const float*)d_in[4];
    const float* W1    = (const float*)d_in[5];
    const float* b1    = (const float*)d_in[6];
    const float* W2    = (const float*)d_in[7];
    const float* b2    = (const float*)d_in[8];
    float*       out   = (float*)d_out;

    const int N = in_sizes[2];          // 50000 nodes
    const int E = in_sizes[1] / 2;      // 800000 edges
    const int* erow = eidx;             // edge_index[0]
    const int* ecol = eidx + E;         // edge_index[1]

    // carve workspace (256B aligned)
    char* ws = (char*)d_ws;
    size_t off = 0;
    auto carve = [&](size_t bytes) -> char* {
        char* p = ws + off;
        off = (off + bytes + 255) & ~(size_t)255;
        return p;
    };
    float* h     = (float*)carve((size_t)N * HID * sizeof(float));
    float* agg   = (float*)carve((size_t)N * HID * sizeof(float));
    float* deg   = (float*)carve((size_t)N * sizeof(float));
    float* dis   = (float*)carve((size_t)N * sizeof(float));
    float* sum_p = (float*)carve((size_t)N_GRAPHS * HID * sizeof(float));
    float* max_p = (float*)carve((size_t)N_GRAPHS * HID * sizeof(float));
    float* cnt   = (float*)carve((size_t)N_GRAPHS * sizeof(float));
    (void)ws_size; (void)n_in; (void)out_size;

    const int T = 256;

    // 1) embedding GEMM (WMMA bf16)
    k_emb_gemm<<<(N + 15) / 16, T, 0, stream>>>(x, emb_W, emb_b, h, N);

    // 2) degrees -> dis = rsqrt(deg)
    k_deg_init <<<(N + T - 1) / T, T, 0, stream>>>(deg, N);
    k_deg_edges<<<(E + T - 1) / T, T, 0, stream>>>(ecol, deg, E);
    k_dis      <<<(N + T - 1) / T, T, 0, stream>>>(deg, dis, N);

    // 3) 4 propagation iterations
    const int nt32 = N * 32;
    const int et32 = E * 32;
    for (int it = 0; it < 4; ++it) {
        k_prop_init <<<(nt32 + T - 1) / T, T, 0, stream>>>(h, dis, agg, N);
        k_prop_edges<<<(et32 + T - 1) / T, T, 0, stream>>>(erow, ecol, dis, h, agg, E);
        k_update    <<<(nt32 + T - 1) / T, T, 0, stream>>>(h, agg, N, it == 3 ? 1 : 0);
    }

    // 4) pooling
    k_pool_init<<<(N_GRAPHS * HID + T - 1) / T, T, 0, stream>>>(sum_p, max_p, cnt);
    k_pool<<<((long)N * HID + T - 1) / T, T, 0, stream>>>(h, batch, sum_p, max_p, cnt, N);

    // 5) MLP head
    k_mlp<<<N_GRAPHS, H1_DIM, 0, stream>>>(sum_p, max_p, cnt, W1, b1, W2, b2, out);
}
